// LigerFusedLinearSimPOLoss_4621384810880
// MI455X (gfx1250) — matrix operand order repaired and checked by reading, not compiled, not run on hardware
//
#include <hip/hip_runtime.h>
#include <hip/hip_bf16.h>
#include <math.h>

#define IGNORE_INDEX (-100)

// ---- CDNA5 types ------------------------------------------------------------
typedef __attribute__((ext_vector_type(16))) __bf16          v16bf;
typedef __attribute__((ext_vector_type(16))) float           f16v;
typedef __attribute__((ext_vector_type(8)))  float           v8f;
typedef __attribute__((ext_vector_type(16))) unsigned short  us16;
typedef __attribute__((ext_vector_type(8)))  unsigned short  us8;
typedef __attribute__((ext_vector_type(4)))  float           f4;
typedef __attribute__((ext_vector_type(4)))  unsigned int    u32x4;
typedef __attribute__((ext_vector_type(8)))  unsigned int    u32x8;

union BF16x16 { v16bf v; unsigned short u[16]; us8 h[2]; };

// ----------------------------------------------------------------------------
// Kernel 0: one-time fp32 -> bf16 convert of X (M x H). X is re-read ~V/128
// times by the GEMM, so converting once halves that recurring traffic and
// lets the TDM copy A tiles without a VALU pass.
// ----------------------------------------------------------------------------
__global__ __launch_bounds__(256)
void convert_x_bf16(const float* __restrict__ X, unsigned short* __restrict__ Xb,
                    size_t n)
{
    size_t i = ((size_t)blockIdx.x * blockDim.x + threadIdx.x) * 16;
    if (i >= n) return;
    f16v t;
    #pragma unroll
    for (int j = 0; j < 4; ++j) {
        f4 v = *(const f4*)(X + i + 4 * j);
        t[4*j+0] = v[0]; t[4*j+1] = v[1]; t[4*j+2] = v[2]; t[4*j+3] = v[3];
    }
    v16bf b = __builtin_convertvector(t, v16bf);
    *(us16*)(Xb + i) = __builtin_bit_cast(us16, b);
}

// ----------------------------------------------------------------------------
// Kernel 1: tiled GEMM (bf16 WMMA) + per-row online softmax partials.
// Block: 256 threads (8 waves). Tile: 128 M-rows x 128 V-cols, K streamed.
// A tiles (128 x 32 bf16) are DMA'd into LDS by the Tensor Data Mover with
// hardware padding producing an 80B row stride (bank-conflict-free for the
// 16-lane ds_load_b128 fragment reads). Double-buffered, TENSORcnt-synced.
// grid.x = M/128 fast-varying so co-resident blocks share each 2MB weight
// slab in L2; weight streams from HBM exactly once. X (32MB bf16) stays
// L2-resident.
// ----------------------------------------------------------------------------
__global__ __launch_bounds__(256)
void simpo_gemm_stats(const float* __restrict__ W,          // (V, H) fp32
                      const unsigned short* __restrict__ Xb,// (M, H) bf16
                      const int*   __restrict__ tgt,        // (M)
                      float* __restrict__ pmax,             // (M, STR)
                      float* __restrict__ psum,             // (M, STR)
                      float* __restrict__ ptl,              // (M) target logits
                      int H, int stride)
{
    const int mb   = blockIdx.x;          // M chunk of 128 rows
    const int nb   = blockIdx.y;          // V chunk of 128 cols
    const int tid  = threadIdx.x;
    const int wave = tid >> 5;
    const int lane = tid & 31;
    const int half = lane >> 4;
    const int l16  = lane & 15;

    const int M0   = mb * 128;
    const int N0   = nb * 128;
    const int wcol = N0 + wave * 16 + l16;   // logits column == weight row

    // Two A buffers: 128 rows x (32 bf16 data + 8 bf16 pad) = 80B stride
    __shared__ unsigned short sA[2][128 * 40];
    __shared__ float sPM[8][128];
    __shared__ float sPS[8][128];

    const unsigned ldsbase = (unsigned)(size_t)&sA[0][0];

    // TDM issue: 2-D tile (tile_dim0=32 elems of 2B, tile_dim1=128 rows),
    // tensor_dim0_stride = H, pad 4 DWORDs every 16 DWORDs -> 80B LDS rows.
    auto tdm_issue = [&](int buf, int kk) {
        unsigned long long ga =
            (unsigned long long)(size_t)Xb + 2ull * ((size_t)M0 * H + (size_t)kk);
        u32x4 g0;
        g0[0] = 1u;                                    // count=1, gather off
        g0[1] = ldsbase + (unsigned)buf * (128u * 80u);// lds_addr (bytes)
        g0[2] = (unsigned)ga;                          // global_addr lo
        g0[3] = (unsigned)(ga >> 32) | 0x80000000u;    // global_addr hi | type=2
        u32x8 g1;
        g1[0] = (1u << 16)    // data_size = 2B
              | (1u << 20)    // pad_enable
              | (3u << 22)    // pad_interval: 16 DWORDs
              | (3u << 25);   // pad_amount: 4 DWORDs
        g1[1] = ((unsigned)H & 0xFFFFu) << 16;                 // tensor_dim0 lo16
        g1[2] = ((unsigned)H >> 16) | (128u << 16);            // dim0 hi | tensor_dim1 lo
        g1[3] = (32u << 16);                                   // tensor_dim1 hi | tile_dim0
        g1[4] = 128u;                                          // tile_dim1 (tile_dim2=0)
        g1[5] = (unsigned)H;                                   // tensor_dim0_stride lo32
        g1[6] = 0u;
        g1[7] = 0u;
        asm volatile("tensor_load_to_lds %0, %1" :: "s"(g0), "s"(g1) : "memory");
    };

    v8f acc[8];
    #pragma unroll
    for (int i = 0; i < 8; ++i)
        #pragma unroll
        for (int r = 0; r < 8; ++r) acc[i][r] = 0.0f;

    // B fragment source: 16 consecutive K-floats per lane (half -> K 0-15 / 16-31)
    const float* wbase = W + (size_t)wcol * H + half * 16;

    if (wave == 0) tdm_issue(0, 0);                 // prologue

    for (int k0 = 0; k0 < H; k0 += 32) {
        const int cur = (k0 >> 5) & 1;
        if (wave == 0) __builtin_amdgcn_s_wait_tensorcnt(0);
        __syncthreads();                            // buffer 'cur' ready for all
        if (wave == 0 && k0 + 32 < H) tdm_issue(cur ^ 1, k0 + 32);

        // ---- B fragment: weight row 'wcol', K = k0 + 16*half + [0..15] ----
        BF16x16 bf;
        {
            const float* wp = wbase + k0;
            f16v t;
            #pragma unroll
            for (int i = 0; i < 4; ++i) {
                f4 v = *(const f4*)(wp + 4 * i);
                t[4*i+0] = v[0]; t[4*i+1] = v[1]; t[4*i+2] = v[2]; t[4*i+3] = v[3];
            }
            bf.v = __builtin_convertvector(t, v16bf);
            if (k0 + 32 < H) __builtin_prefetch(wp + 32, 0, 1);  // global_prefetch_b8
        }

        // ---- 8 M-subtiles: A fragments from LDS, accumulate with WMMA ----
        #pragma unroll
        for (int mt = 0; mt < 8; ++mt) {
            // A 16x32 bf16 layout: lane m=l16; elems 0-15: K=16*half+0..15
            const unsigned short* ap = &sA[cur][(mt * 16 + l16) * 40];
            BF16x16 af;
            af.h[0] = *(const us8*)(ap + 16 * half);
            af.h[1] = *(const us8*)(ap + 16 * half + 8);
            acc[mt] = __builtin_amdgcn_wmma_f32_16x16x32_bf16(
                false, af.v, false, bf.v, (short)0, acc[mt], false, false);
        }
    }

    // ---- per-row reduction over this block's 128 columns ----
    // C layout: lane N = l16, VGPR r -> M = mt*16 + r + 8*half
    #pragma unroll
    for (int mt = 0; mt < 8; ++mt) {
        #pragma unroll
        for (int r = 0; r < 8; ++r) {
            float v    = acc[mt][r];
            int   rowl = mt * 16 + r + 8 * half;
            int   rowg = M0 + rowl;
            if (tgt[rowg] == wcol) ptl[rowg] = v;   // unique writer per row
            float m = v;
            #pragma unroll
            for (int d = 1; d < 16; d <<= 1) m = fmaxf(m, __shfl_xor(m, d, 32));
            float s = __expf(v - m);
            #pragma unroll
            for (int d = 1; d < 16; d <<= 1) s += __shfl_xor(s, d, 32);
            if (l16 == 0) { sPM[wave][rowl] = m; sPS[wave][rowl] = s; }
        }
    }
    __syncthreads();
    if (tid < 128) {
        float gm = -INFINITY, gs = 0.0f;
        #pragma unroll
        for (int wv = 0; wv < 8; ++wv) {
            float m  = sPM[wv][tid], s = sPS[wv][tid];
            float nm = fmaxf(gm, m);
            gs = gs * __expf(gm - nm) + s * __expf(m - nm);
            gm = nm;
        }
        int rowg = M0 + tid;
        pmax[(size_t)rowg * stride + nb] = gm;
        psum[(size_t)rowg * stride + nb] = gs;
    }
}

// ----------------------------------------------------------------------------
// Kernel 2: combine per-chunk partials -> token log-probs
// ----------------------------------------------------------------------------
__global__ void simpo_rowcombine(const float* __restrict__ pmax,
                                 const float* __restrict__ psum,
                                 const float* __restrict__ ptl,
                                 const int*   __restrict__ tgt,
                                 float* __restrict__ tok,
                                 int nchunks, int stride, int M)
{
    int row = blockIdx.x * blockDim.x + threadIdx.x;
    if (row >= M) return;
    float gm = -INFINITY, gs = 0.0f;
    for (int c = 0; c < nchunks; ++c) {
        float m  = pmax[(size_t)row * stride + c];
        float s  = psum[(size_t)row * stride + c];
        float nm = fmaxf(gm, m);
        gs = gs * __expf(gm - nm) + s * __expf(m - nm);
        gm = nm;
    }
    float t = (tgt[row] == IGNORE_INDEX) ? 0.0f
                                         : ptl[row] - (gm + __logf(gs));
    tok[row] = t;
}

// ----------------------------------------------------------------------------
// Kernel 3: masked per-sequence average of token log-probs
// ----------------------------------------------------------------------------
__global__ __launch_bounds__(256)
void simpo_seqavg(const float* __restrict__ tok, const int* __restrict__ tgt,
                  float* __restrict__ avg, int T)
{
    int b = blockIdx.x, tid = threadIdx.x;
    float s = 0.0f, c = 0.0f;
    for (int t = tid; t < T; t += 256) {
        int   rowg = b * T + t;
        float m    = (tgt[rowg] != IGNORE_INDEX) ? 1.0f : 0.0f;
        s += tok[rowg] * m;
        c += m;
    }
    #pragma unroll
    for (int d = 1; d < 32; d <<= 1) { s += __shfl_xor(s, d, 32); c += __shfl_xor(c, d, 32); }
    __shared__ float ss[8], sc[8];
    if ((tid & 31) == 0) { ss[tid >> 5] = s; sc[tid >> 5] = c; }
    __syncthreads();
    if (tid == 0) {
        float S = 0.0f, C = 0.0f;
        for (int i = 0; i < 8; ++i) { S += ss[i]; C += sc[i]; }
        avg[b] = S / fmaxf(C, 1.0f);
    }
}

// ----------------------------------------------------------------------------
// Kernel 4: SimPO pairwise loss (label smoothing = 0)
// ----------------------------------------------------------------------------
__global__ void simpo_loss(const float* __restrict__ avg, float* __restrict__ out,
                           int npairs, float beta, float gamma)
{
    if (blockIdx.x == 0 && threadIdx.x == 0) {
        float acc = 0.0f;
        for (int i = 0; i < npairs; ++i) {
            float z = beta * (avg[i] - avg[i + npairs]) - gamma;
            // -log_sigmoid(z) = softplus(-z), numerically stable
            acc += fmaxf(-z, 0.0f) + log1pf(__expf(-fabsf(z)));
        }
        out[0] = acc / (float)npairs;
    }
}

// ----------------------------------------------------------------------------
extern "C" void kernel_launch(void* const* d_in, const int* in_sizes, int n_in,
                              void* d_out, int out_size, void* d_ws, size_t ws_size,
                              hipStream_t stream)
{
    const float* W   = (const float*)d_in[0];  // (V, H) fp32
    const float* X   = (const float*)d_in[1];  // (B, T, H) fp32
    const int*   tgt = (const int*)d_in[2];    // (B, T)
    float*       out = (float*)d_out;

    const int M = in_sizes[2];          // B*T = 4096
    const int H = in_sizes[1] / M;      // 4096
    const int V = in_sizes[0] / H;      // 32000
    const int T = 512, B = 8;           // reference-fixed shape
    const int NCH = V / 128;            // 250 column chunks
    const int STR = 256;                // padded chunk stride

    unsigned short* Xb = (unsigned short*)d_ws;     // M * H bf16
    float* pmax = (float*)(Xb + (size_t)M * H);     // M * STR
    float* psum = pmax + (size_t)M * STR;           // M * STR
    float* ptl  = psum + (size_t)M * STR;           // M
    float* tok  = ptl + M;                          // M
    float* avg  = tok + M;                          // B

    const size_t nX = (size_t)M * H;
    convert_x_bf16<<<(unsigned)((nX / 16 + 255) / 256), 256, 0, stream>>>(X, Xb, nX);

    dim3 g1(M / 128, NCH);
    simpo_gemm_stats<<<g1, 256, 0, stream>>>(W, Xb, tgt, pmax, psum, ptl, H, STR);
    simpo_rowcombine<<<(M + 255) / 256, 256, 0, stream>>>(pmax, psum, ptl, tgt, tok, NCH, STR, M);
    simpo_seqavg<<<B, 256, 0, stream>>>(tok, tgt, avg, T);
    simpo_loss<<<1, 32, 0, stream>>>(avg, out, B / 2, 0.1f, 0.5f);
}
